// ContrastiveLoss_26920855012068
// MI455X (gfx1250) — matrix operand order repaired and checked by reading, not compiled, not run on hardware
//
#include <hip/hip_runtime.h>
#include <hip/hip_bf16.h>

// ---------------------------------------------------------------------------
// ContrastiveLoss (NT-Xent w/ euclidean metric), B=4096, D=256, T=0.1
// Fused:  normalize -> split-bf16 WMMA GEMM tile + exp + row-reduce -> loss
// Never materializes the 8192x8192 sim matrix (saves ~512MB of HBM traffic).
// ---------------------------------------------------------------------------

typedef __attribute__((ext_vector_type(16))) __bf16 v16bf;
typedef __attribute__((ext_vector_type(8)))  float  v8f;

#define NB    4096
#define ND    256
#define NR    8192           // 2*B rows
#define INV_T 10.0f

// bf16 round-to-nearest-even from f32
__device__ inline unsigned short f2bf(float x) {
  unsigned u = __float_as_uint(x);
  unsigned r = u + 0x7FFFu + ((u >> 16) & 1u);
  return (unsigned short)(r >> 16);
}
__device__ inline float bf2f(unsigned short h) {
  return __uint_as_float(((unsigned)h) << 16);
}

union FragBF { uint4 u[2]; v16bf v; };

// ---------------------------------------------------------------------------
// Kernel 1: L2-normalize rows, emit split-bf16 (hi + residual lo), sq[r],
// and zero the denom accumulator. One wave (32 lanes) per row, 8 f32/lane.
// ---------------------------------------------------------------------------
__global__ __launch_bounds__(256) void nt_normalize(
    const float* __restrict__ emb_i, const float* __restrict__ emb_j,
    unsigned short* __restrict__ zhi, unsigned short* __restrict__ zlo,
    float* __restrict__ sq, float* __restrict__ denom)
{
  int gid = blockIdx.x * blockDim.x + threadIdx.x;
  if (gid < NR) denom[gid] = 0.0f;          // zero accumulator (stream-ordered)

  const int lane = threadIdx.x & 31;
  const int wave = threadIdx.x >> 5;
  const int r = blockIdx.x * 8 + wave;      // 0..8191
  const float* src = (r < NB) ? (emb_i + (size_t)r * ND)
                              : (emb_j + (size_t)(r - NB) * ND);
  const float4* p4 = (const float4*)(src + lane * 8);
  float4 a = p4[0], b = p4[1];
  float x[8] = {a.x, a.y, a.z, a.w, b.x, b.y, b.z, b.w};

  float ss = 0.0f;
  #pragma unroll
  for (int i = 0; i < 8; ++i) ss += x[i] * x[i];
  #pragma unroll
  for (int m = 16; m >= 1; m >>= 1) ss += __shfl_xor(ss, m, 32);

  float n   = sqrtf(ss);
  float dn  = fmaxf(n, 1e-12f);             // matches F.normalize semantics
  float inv = 1.0f / dn;
  float sqv = ss * inv * inv;               // sum(z^2)

  unsigned short h[8], l[8];
  #pragma unroll
  for (int i = 0; i < 8; ++i) {
    float z = x[i] * inv;
    h[i] = f2bf(z);
    l[i] = f2bf(z - bf2f(h[i]));            // residual: hi+lo ~ fp32-accurate
  }
  uint4 uh, ul;
  uh.x = (unsigned)h[0] | ((unsigned)h[1] << 16);
  uh.y = (unsigned)h[2] | ((unsigned)h[3] << 16);
  uh.z = (unsigned)h[4] | ((unsigned)h[5] << 16);
  uh.w = (unsigned)h[6] | ((unsigned)h[7] << 16);
  ul.x = (unsigned)l[0] | ((unsigned)l[1] << 16);
  ul.y = (unsigned)l[2] | ((unsigned)l[3] << 16);
  ul.z = (unsigned)l[4] | ((unsigned)l[5] << 16);
  ul.w = (unsigned)l[6] | ((unsigned)l[7] << 16);
  *(uint4*)(zhi + (size_t)r * ND + lane * 8) = uh;
  *(uint4*)(zlo + (size_t)r * ND + lane * 8) = ul;
  if (lane == 0) sq[r] = sqv;
}

// ---------------------------------------------------------------------------
// Kernel 2: fused S = Z Z^T tile (128x128) via split-bf16 WMMA, then
// sim = -(sq_r + sq_c - 2*dot); denom[r] += (r!=c) * exp(sim/T).
// 8 waves/block; wave w owns 16-row strip x 8 col tiles (8x v8f acc).
// Operands read straight from global (8MB working set is L2-resident).
// A-fragment layout (16-bit 16x32): lane=16*half+m; elems 0..15 <- K=16*half+[0,16).
//   B-fragment (32x16): lane=16*half+n;
//   elems 0..15 <- K=16*half+[0,16).  Both are contiguous 16B-aligned loads.
// ---------------------------------------------------------------------------
__global__ __launch_bounds__(256) void nt_sim_rowsum(
    const unsigned short* __restrict__ zhi, const unsigned short* __restrict__ zlo,
    const float* __restrict__ sq, float* __restrict__ denom)
{
  const int lane = threadIdx.x & 31;
  const int wave = threadIdx.x >> 5;
  const int half = lane >> 4;
  const int ln   = lane & 15;
  const int r0 = blockIdx.y * 128;
  const int c0 = blockIdx.x * 128;
  const int rb = r0 + wave * 16;            // this wave's 16-row strip
  const int m  = rb + ln;                   // A row held by this lane

  v8f acc[8];
  #pragma unroll
  for (int t = 0; t < 8; ++t)
    #pragma unroll
    for (int j = 0; j < 8; ++j) acc[t][j] = 0.0f;

  const unsigned short* arow_h = zhi + (size_t)m * ND;
  const unsigned short* arow_l = zlo + (size_t)m * ND;

  for (int kk = 0; kk < ND; kk += 32) {
    FragBF ahi, alo;
    ahi.u[0] = *(const uint4*)(arow_h + kk + 16 * half);
    ahi.u[1] = *(const uint4*)(arow_h + kk + 16 * half + 8);
    alo.u[0] = *(const uint4*)(arow_l + kk + 16 * half);
    alo.u[1] = *(const uint4*)(arow_l + kk + 16 * half + 8);

    #pragma unroll
    for (int ct = 0; ct < 8; ++ct) {
      const int c = c0 + ct * 16 + ln;      // B column = Z row index
      const unsigned short* bh = zhi + (size_t)c * ND + kk + 16 * half;
      const unsigned short* bl = zlo + (size_t)c * ND + kk + 16 * half;
      FragBF bhi, blo;
      bhi.u[0] = *(const uint4*)(bh);
      bhi.u[1] = *(const uint4*)(bh + 8);
      blo.u[0] = *(const uint4*)(bl);
      blo.u[1] = *(const uint4*)(bl + 8);
      // dot ~= hi*hi + hi*lo + lo*hi  (fp32 accumulate)
      acc[ct] = __builtin_amdgcn_wmma_f32_16x16x32_bf16(
          false, ahi.v, false, bhi.v, (short)0, acc[ct], false, false);
      acc[ct] = __builtin_amdgcn_wmma_f32_16x16x32_bf16(
          false, ahi.v, false, blo.v, (short)0, acc[ct], false, false);
      acc[ct] = __builtin_amdgcn_wmma_f32_16x16x32_bf16(
          false, alo.v, false, bhi.v, (short)0, acc[ct], false, false);
    }
  }

  // Epilogue: D layout -> VGPR p holds row (rb + p + 8*half), col c0+ct*16+ln
  float rowsum[8];
  #pragma unroll
  for (int p = 0; p < 8; ++p) {
    const int r  = rb + p + 8 * half;
    const float sqr = sq[r];
    float s = 0.0f;
    #pragma unroll
    for (int ct = 0; ct < 8; ++ct) {
      const int c = c0 + ct * 16 + ln;
      float sim = -(sqr + sq[c] - 2.0f * acc[ct][p]);
      s += (r == c) ? 0.0f : __expf(sim * INV_T);
    }
    rowsum[p] = s;
  }
  // reduce over the 16 lanes sharing each row (xor masks stay within half)
  #pragma unroll
  for (int p = 0; p < 8; ++p) {
    float v = rowsum[p];
    #pragma unroll
    for (int msk = 1; msk < 16; msk <<= 1) v += __shfl_xor(v, msk, 32);
    rowsum[p] = v;
  }
  if (ln == 0) {
    #pragma unroll
    for (int p = 0; p < 8; ++p)
      atomicAdd(&denom[rb + p + 8 * half], rowsum[p]);   // global_atomic_add_f32
  }
}

// ---------------------------------------------------------------------------
// Kernel 3: positives  pos[i] = sim[i, i+B] = -(sq_i + sq_{i+B} - 2 z_i.z_j)
// (symmetric, so sim_ji uses the same value). One wave per row.
// ---------------------------------------------------------------------------
__global__ __launch_bounds__(256) void nt_positives(
    const unsigned short* __restrict__ zhi, const unsigned short* __restrict__ zlo,
    const float* __restrict__ sq, float* __restrict__ pos)
{
  const int lane = threadIdx.x & 31;
  const int wave = threadIdx.x >> 5;
  const int i = blockIdx.x * 8 + wave;      // 0..4095
  const unsigned short* ih = zhi + (size_t)i * ND + lane * 8;
  const unsigned short* il = zlo + (size_t)i * ND + lane * 8;
  const unsigned short* jh = zhi + (size_t)(i + NB) * ND + lane * 8;
  const unsigned short* jl = zlo + (size_t)(i + NB) * ND + lane * 8;
  float d = 0.0f;
  #pragma unroll
  for (int e = 0; e < 8; ++e) {
    float zi = bf2f(ih[e]) + bf2f(il[e]);
    float zj = bf2f(jh[e]) + bf2f(jl[e]);
    d += zi * zj;
  }
  #pragma unroll
  for (int m = 16; m >= 1; m >>= 1) d += __shfl_xor(d, m, 32);
  if (lane == 0) pos[i] = -(sq[i] + sq[i + NB] - 2.0f * d);
}

// ---------------------------------------------------------------------------
// Kernel 4: loss = mean_r( -pos[r]/T + log(denom[r]) )
// ---------------------------------------------------------------------------
__global__ __launch_bounds__(256) void nt_loss(
    const float* __restrict__ denom, const float* __restrict__ pos,
    float* __restrict__ out)
{
  __shared__ float red[256];
  const int t = threadIdx.x;
  float s = 0.0f;
  for (int r = t; r < NR; r += 256) {
    float p = pos[(r < NB) ? r : (r - NB)];
    s += -INV_T * p + __logf(denom[r]);
  }
  red[t] = s;
  __syncthreads();
  for (int st = 128; st > 0; st >>= 1) {
    if (t < st) red[t] += red[t + st];
    __syncthreads();
  }
  if (t == 0) out[0] = red[0] * (1.0f / (float)NR);
}

// ---------------------------------------------------------------------------
extern "C" void kernel_launch(void* const* d_in, const int* in_sizes, int n_in,
                              void* d_out, int out_size, void* d_ws, size_t ws_size,
                              hipStream_t stream) {
  const float* emb_i = (const float*)d_in[0];
  const float* emb_j = (const float*)d_in[1];
  float* out = (float*)d_out;

  // workspace layout (needs ~8.1 MB)
  char* ws = (char*)d_ws;
  unsigned short* zhi   = (unsigned short*)(ws);                       // 4 MB
  unsigned short* zlo   = (unsigned short*)(ws + (size_t)NR * ND * 2); // 4 MB
  float*          sq    = (float*)(ws + (size_t)NR * ND * 4);          // 32 KB
  float*          denom = (float*)(ws + (size_t)NR * ND * 4 + 32768);  // 32 KB
  float*          pos   = (float*)(ws + (size_t)NR * ND * 4 + 65536);  // 16 KB

  nt_normalize <<<NR / 8, 256, 0, stream>>>(emb_i, emb_j, zhi, zlo, sq, denom);
  nt_sim_rowsum<<<dim3(NR / 128, NR / 128), 256, 0, stream>>>(zhi, zlo, sq, denom);
  nt_positives <<<NB / 8, 256, 0, stream>>>(zhi, zlo, sq, pos);
  nt_loss      <<<1, 256, 0, stream>>>(denom, pos, out);
}
